// MultiHeadAttentionClassical_65481071401274
// MI455X (gfx1250) — compile-verified, matrix operand order checked by reading, not run on hardware
//
#include <hip/hip_runtime.h>
#include <hip/hip_bf16.h>

// ---------------------------------------------------------------------------
// MultiHeadAttention (classical) for MI455X / gfx1250, wave32 + WMMA f16.
//   B=2, S=2048, E=1024, H=16, Dk=64.  M_total = B*S = 4096 rows.
// Pipeline: f32->f16 cast -> register-blocked WMMA GEMMs (Q,K,V^T) ->
// flash-attention (async-to-LDS double-buffered K/V staging, WMMA scores,
// online softmax: DPP row-max + WMMA row-sum, LDS P transpose) -> WMMA proj.
// ---------------------------------------------------------------------------

typedef __attribute__((ext_vector_type(16))) _Float16 v16h;
typedef __attribute__((ext_vector_type(8)))  float    v8f;

#define B_SZ   2
#define S_SZ   2048
#define E_SZ   1024
#define H_SZ   16
#define DK_SZ  64
#define MTOT   (B_SZ * S_SZ)            // 4096

// ---------------------------------------------------------------------------
// Load one 16x32 f16 A/B fragment (v16h) for v_wmma_f32_16x16x32_f16.
// Row-major [*, ld] f16 source; per ISA 7.12.2: lanes 0-15 hold K=k0+0..7 then
// k0+16..23; lanes 16-31 hold K=k0+8..15 then k0+24..31.  Two 16B loads/lane.
// ---------------------------------------------------------------------------
__device__ inline v16h load_frag16(const _Float16* p, int ld, int row, int k0,
                                   int lane) {
  const _Float16* base = p + (size_t)row * ld + k0 + ((lane >> 4) << 3);
  union { v16h h; uint4 q[2]; } f;
  f.q[0] = *(const uint4*)(base);
  f.q[1] = *(const uint4*)(base + 16);
  return f.h;
}

__device__ inline v8f wmma_f16(v16h a, v16h b, v8f c) {
  return __builtin_amdgcn_wmma_f32_16x16x32_f16(false, a, false, b, (short)0,
                                                c, false, false);
}

// Async DMA of one 16B chunk per lane into LDS (gfx1250, tracked by ASYNCcnt).
// lds_off = low 32 bits of the generic pointer (LDS aperture keeps addr[31:0]).
__device__ inline void async_copy_b128(const void* lds_dst, const void* gsrc) {
  uint32_t lds_off = (uint32_t)(size_t)lds_dst;
  asm volatile("global_load_async_to_lds_b128 %0, %1, off"
               :: "v"(lds_off), "v"(gsrc)
               : "memory");
}

// All-lane max over each 16-lane half of a wave32 using DPP row rotates
// (v_mov_b32_dpp row_ror:{1,2,4,8} + v_max) -- pure VALU, no LDS traffic.
template <int CTRL>
__device__ inline float fmax_dpp16(float x) {
  int xi = __builtin_bit_cast(int, x);
  int yi = __builtin_amdgcn_update_dpp(xi, xi, CTRL, 0xf, 0xf, true);
  return fmaxf(x, __builtin_bit_cast(float, yi));
}
__device__ inline float rowmax16(float x) {
  x = fmax_dpp16<0x121>(x);  // row_ror:1
  x = fmax_dpp16<0x122>(x);  // row_ror:2
  x = fmax_dpp16<0x124>(x);  // row_ror:4
  x = fmax_dpp16<0x128>(x);  // row_ror:8
  return x;
}

// ---------------------------------------------------------------------------
// f32 -> f16 cast, 4 elements/thread.
// ---------------------------------------------------------------------------
__global__ void cvt_f32_to_f16(const float* __restrict__ in,
                               _Float16* __restrict__ out, int n4) {
  int i = blockIdx.x * blockDim.x + threadIdx.x;
  if (i < n4) {
    float4 v = ((const float4*)in)[i];
    union { _Float16 h[4]; uint2 u; } r;
    r.h[0] = (_Float16)v.x; r.h[1] = (_Float16)v.y;
    r.h[2] = (_Float16)v.z; r.h[3] = (_Float16)v.w;
    ((uint2*)out)[i] = r.u;
  }
}

// ---------------------------------------------------------------------------
// GEMM: Out[M,N] = A[M,K] * Bw[N,K]^T, register-blocked: one wave32 owns a
// 16x64 output tile (4 C accumulators). Per k-step: 1 A frag + 4 B frags ->
// 4 WMMAs (4x A reuse). OUT_F32 writes f32 (final proj); TRANS writes Out^T.
// ---------------------------------------------------------------------------
template <bool OUT_F32, bool TRANS>
__global__ __launch_bounds__(256) void gemm16x64_wmma(
    const _Float16* __restrict__ A, const _Float16* __restrict__ Bw,
    void* __restrict__ Out, int M, int N, int K, int ldo) {
  const int lane = threadIdx.x & 31;
  const int wid  = blockIdx.x * (blockDim.x >> 5) + (threadIdx.x >> 5);
  const int tiles_n = N >> 6;                 // 64-wide N tiles
  const int tm = wid / tiles_n;
  const int tn = wid % tiles_n;
  if (tm >= (M >> 4)) return;                 // wave-uniform

  const int ra = (tm << 4) + (lane & 15);
  const int rb = (tn << 6) + (lane & 15);

  v8f c0 = {}, c1 = {}, c2 = {}, c3 = {};
#pragma unroll 2
  for (int k0 = 0; k0 < K; k0 += 32) {
    v16h a  = load_frag16(A,  K, ra,      k0, lane);
    v16h b0 = load_frag16(Bw, K, rb,      k0, lane);
    v16h b1 = load_frag16(Bw, K, rb + 16, k0, lane);
    v16h b2 = load_frag16(Bw, K, rb + 32, k0, lane);
    v16h b3 = load_frag16(Bw, K, rb + 48, k0, lane);
    c0 = wmma_f16(a, b0, c0);
    c1 = wmma_f16(a, b1, c1);
    c2 = wmma_f16(a, b2, c2);
    c3 = wmma_f16(a, b3, c3);
  }

  const int hi8 = (lane >> 4) << 3;
  v8f acc[4] = {c0, c1, c2, c3};
#pragma unroll
  for (int t = 0; t < 4; ++t) {
    const int n = (tn << 6) + (t << 4) + (lane & 15);
#pragma unroll
    for (int r = 0; r < 8; ++r) {
      const int row = (tm << 4) + r + hi8;
      if constexpr (OUT_F32) {
        ((float*)Out)[(size_t)row * ldo + n] = acc[t][r];
      } else if constexpr (TRANS) {
        ((_Float16*)Out)[(size_t)n * ldo + row] = (_Float16)acc[t][r];
      } else {
        ((_Float16*)Out)[(size_t)row * ldo + n] = (_Float16)acc[t][r];
      }
    }
  }
}

// ---------------------------------------------------------------------------
// Flash attention. Block = 8 waves sharing one (batch, head); wave ws handles
// query tile qt0+ws. K/V chunks (32 keys) are staged into LDS with
// global_load_async_to_lds_b128, double-buffered, drained by s_wait_asynccnt.
// Row maxes via DPP rotates (VALU only); row sums via a 5th WMMA against an
// all-ones B fragment (same alpha recurrence as the O accumulators).
// 1/sqrt(dk)=0.125 is exact in f16 and pre-folded into the Q fragments.
// ---------------------------------------------------------------------------
__global__ __launch_bounds__(256) void flash_attn_wmma(
    const _Float16* __restrict__ Qh, const _Float16* __restrict__ Kh,
    const _Float16* __restrict__ Vt, _Float16* __restrict__ Oh) {
  __shared__ __attribute__((aligned(16))) _Float16 kbuf[2][32][64];  // 2x4KB
  __shared__ __attribute__((aligned(16))) _Float16 vbuf[2][64][32];  // 2x4KB
  __shared__ __attribute__((aligned(16))) _Float16 plds[8][16][32];  // 8KB

  const int tid  = threadIdx.x;
  const int lane = tid & 31;
  const int ws   = tid >> 5;

  // (b, h) are block-uniform; 8 waves cover 8 consecutive query tiles.
  const int w0 = blockIdx.x * 8;
  const int b  = w0 >> 11;
  const int h  = (w0 >> 7) & 15;
  const int qt = (w0 & 127) + ws;

  const _Float16* Qb = Qh + (((size_t)(b * S_SZ + (qt << 4))) << 10) + (h << 6);
  const _Float16* Kb = Kh + (((size_t)(b * S_SZ)) << 10) + (h << 6);
  const _Float16* Vb = Vt + (((size_t)(h << 6)) << 12) + b * S_SZ;  // Vt[d][m]

  // Cooperative async staging of one 32-key chunk: K rows are 128B (8 lanes),
  // V rows are 64B (4 lanes); 256 threads move 4KB+4KB with 2 asyncs/wave.
  auto stage_chunk = [&](int kc, int buf) {
    const _Float16* kg = Kb + (size_t)(kc + (tid >> 3)) * E_SZ + ((tid & 7) << 3);
    async_copy_b128(&kbuf[buf][tid >> 3][(tid & 7) << 3], kg);
    const _Float16* vg = Vb + (size_t)(tid >> 2) * MTOT + kc + ((tid & 3) << 3);
    async_copy_b128(&vbuf[buf][tid >> 2][(tid & 3) << 3], vg);
  };

  // Q fragments resident for the whole key loop, pre-scaled by 1/sqrt(dk).
  v16h qscale;
#pragma unroll
  for (int i = 0; i < 16; ++i) qscale[i] = (_Float16)0.125f;
  const v16h aq0 = load_frag16(Qb, E_SZ, lane & 15, 0,  lane) * qscale;
  const v16h aq1 = load_frag16(Qb, E_SZ, lane & 15, 32, lane) * qscale;

  v16h ones;
#pragma unroll
  for (int i = 0; i < 16; ++i) ones[i] = (_Float16)1.0f;

  v8f o0 = {}, o1 = {}, o2 = {}, o3 = {}, osum = {};
  float mi[8];
#pragma unroll
  for (int r = 0; r < 8; ++r) mi[r] = -1e30f;

  constexpr int NCHUNK = S_SZ / 32;  // 64

  stage_chunk(0, 0);

  for (int i = 0; i < NCHUNK; ++i) {
    const int cur = i & 1;
    if (i + 1 < NCHUNK) {
      stage_chunk((i + 1) * 32, cur ^ 1);
      asm volatile("s_wait_asynccnt 0x2" ::: "memory");  // chunk i landed
    } else {
      asm volatile("s_wait_asynccnt 0x0" ::: "memory");
    }
    __syncthreads();  // all waves' slices of chunk i visible

    // ---- scores S = (Q/8) K^T for 32 staged keys (two 16-key N tiles) ----
    const _Float16* kl = &kbuf[cur][0][0];
    v16h bk00 = load_frag16(kl, 64,      (lane & 15), 0,  lane);
    v16h bk01 = load_frag16(kl, 64,      (lane & 15), 32, lane);
    v16h bk10 = load_frag16(kl, 64, 16 + (lane & 15), 0,  lane);
    v16h bk11 = load_frag16(kl, 64, 16 + (lane & 15), 32, lane);
    v8f c0 = {}, c1 = {};
    c0 = wmma_f16(aq0, bk00, c0); c0 = wmma_f16(aq1, bk01, c0);
    c1 = wmma_f16(aq0, bk10, c1); c1 = wmma_f16(aq1, bk11, c1);

    // ---- online softmax (C layout: vgpr r, M = r + 8*(lane>=16)) ----
#pragma unroll
    for (int r = 0; r < 8; ++r) {
      const float s0 = c0[r];
      const float s1 = c1[r];
      const float t = rowmax16(fmaxf(s0, s1));   // DPP, no LDS
      const float mnew  = fmaxf(mi[r], t);
      const float alpha = __expf(mi[r] - mnew);
      mi[r] = mnew;
      const float p0 = __expf(s0 - mnew);
      const float p1 = __expf(s1 - mnew);
      o0[r] *= alpha; o1[r] *= alpha; o2[r] *= alpha; o3[r] *= alpha;
      osum[r] *= alpha;  // row-sum accumulator obeys the same recurrence
      const int row = r + ((lane >> 4) << 3);
      plds[ws][row][lane & 15]        = (_Float16)p0;
      plds[ws][row][16 + (lane & 15)] = (_Float16)p1;
    }

    // wave-private LDS tile: DS-counter drain only, no barrier
    asm volatile("s_wait_dscnt 0x0" ::: "memory");

    // ---- P reloaded as an A fragment (16 q x 32 keys) ----
    v16h ap = load_frag16(&plds[ws][0][0], 32, lane & 15, 0, lane);

    // ---- O += P V and rowsum += P * ones, from staged V^T tile ----
    const _Float16* vl = &vbuf[cur][0][0];
    v16h bv0 = load_frag16(vl, 32,      (lane & 15), 0, lane);
    v16h bv1 = load_frag16(vl, 32, 16 + (lane & 15), 0, lane);
    v16h bv2 = load_frag16(vl, 32, 32 + (lane & 15), 0, lane);
    v16h bv3 = load_frag16(vl, 32, 48 + (lane & 15), 0, lane);
    o0 = wmma_f16(ap, bv0, o0);
    o1 = wmma_f16(ap, bv1, o1);
    o2 = wmma_f16(ap, bv2, o2);
    o3 = wmma_f16(ap, bv3, o3);
    osum = wmma_f16(ap, ones, osum);   // per-row sum of P, alpha-consistent

    __syncthreads();  // everyone done reading chunk i before it is overwritten
  }

  // ---- finalize: divide by row sums, store 16x64 f16 tile ----
  const int n = lane & 15;
#pragma unroll
  for (int r = 0; r < 8; ++r) {
    const float inv = 1.0f / osum[r];
    const int row = b * S_SZ + (qt << 4) + r + ((lane >> 4) << 3);
    _Float16* op = Oh + ((size_t)row << 10) + (h << 6) + n;
    op[0]  = (_Float16)(o0[r] * inv);
    op[16] = (_Float16)(o1[r] * inv);
    op[32] = (_Float16)(o2[r] * inv);
    op[48] = (_Float16)(o3[r] * inv);
  }
}

// ---------------------------------------------------------------------------
// Launch: cast -> Q/K/V^T GEMMs -> flash attention -> output projection.
// Workspace (f16 elems): xh 4M | W*4 1M each | Q 4M | K 4M | Vt 4M | O 4M
// => 24M halves = 48 MB.
// ---------------------------------------------------------------------------
extern "C" void kernel_launch(void* const* d_in, const int* in_sizes, int n_in,
                              void* d_out, int out_size, void* d_ws,
                              size_t ws_size, hipStream_t stream) {
  const float* x  = (const float*)d_in[0];
  const float* Wq = (const float*)d_in[1];
  const float* Wk = (const float*)d_in[2];
  const float* Wv = (const float*)d_in[3];
  const float* Wo = (const float*)d_in[4];
  float* out = (float*)d_out;

  const size_t XN = (size_t)MTOT * E_SZ;   // 4M
  const size_t WN = (size_t)E_SZ * E_SZ;   // 1M

  _Float16* xh  = (_Float16*)d_ws;
  _Float16* wqh = xh  + XN;
  _Float16* wkh = wqh + WN;
  _Float16* wvh = wkh + WN;
  _Float16* woh = wvh + WN;
  _Float16* qh  = woh + WN;
  _Float16* kh  = qh  + XN;
  _Float16* vt  = kh  + XN;
  _Float16* oh  = vt  + XN;

  cvt_f32_to_f16<<<(int)(XN / 4 / 256), 256, 0, stream>>>(x,  xh,  (int)(XN / 4));
  cvt_f32_to_f16<<<(int)(WN / 4 / 256), 256, 0, stream>>>(Wq, wqh, (int)(WN / 4));
  cvt_f32_to_f16<<<(int)(WN / 4 / 256), 256, 0, stream>>>(Wk, wkh, (int)(WN / 4));
  cvt_f32_to_f16<<<(int)(WN / 4 / 256), 256, 0, stream>>>(Wv, wvh, (int)(WN / 4));
  cvt_f32_to_f16<<<(int)(WN / 4 / 256), 256, 0, stream>>>(Wo, woh, (int)(WN / 4));

  // (4096/16)*(1024/64) = 4096 waves / 8 per block = 512 blocks
  const int gemm_blocks = (MTOT / 16) * (E_SZ / 64) / 8;
  gemm16x64_wmma<false, false><<<gemm_blocks, 256, 0, stream>>>(
      xh, wqh, (void*)qh, MTOT, E_SZ, E_SZ, E_SZ);
  gemm16x64_wmma<false, false><<<gemm_blocks, 256, 0, stream>>>(
      xh, wkh, (void*)kh, MTOT, E_SZ, E_SZ, E_SZ);
  gemm16x64_wmma<false, true><<<gemm_blocks, 256, 0, stream>>>(
      xh, wvh, (void*)vt, MTOT, E_SZ, E_SZ, MTOT);  // V^T: [E][MTOT]

  flash_attn_wmma<<<512, 256, 0, stream>>>(qh, kh, vt, oh);

  gemm16x64_wmma<true, false><<<gemm_blocks, 256, 0, stream>>>(
      oh, woh, (void*)out, MTOT, E_SZ, E_SZ, E_SZ);
}
